// EdgeConvBlock_25340307046480
// MI455X (gfx1250) — matrix operand order repaired and checked by reading, not compiled, not run on hardware
//
#include <hip/hip_runtime.h>
#include <hip/hip_bf16.h>
#include <math.h>
#include <stdint.h>

typedef float  v2f  __attribute__((ext_vector_type(2)));
typedef float  v8f  __attribute__((ext_vector_type(8)));
typedef __bf16 bf16;
typedef bf16   v16bf __attribute__((ext_vector_type(16)));

#define BATCH   8
#define NPTS    4096
#define KNB     20
#define CIN     64
#define COUT    128
#define TOTAL   (BATCH * NPTS)        // 32768 points
#define PG      4                     // points per MLP block
#define ROWS    (PG * KNB)            // 80 edge rows per MLP block (5 M-tiles)

// ---------------------------------------------------------------------------
// Kernel 0: per-point squared norm
// ---------------------------------------------------------------------------
__global__ void sq_kernel(const float* __restrict__ x, float* __restrict__ sq) {
    int i = blockIdx.x * blockDim.x + threadIdx.x;
    if (i >= TOTAL) return;
    const float* r = x + (size_t)i * CIN;
    float s = 0.f;
#pragma unroll
    for (int c = 0; c < CIN; ++c) s += r[c] * r[c];
    sq[i] = s;
}

// ---------------------------------------------------------------------------
// Kernel 0b: convert + transpose weights to bf16 [out][in] for WMMA B-frags
// ---------------------------------------------------------------------------
__global__ void prep_weights(const float* __restrict__ W1, const float* __restrict__ W2,
                             bf16* __restrict__ Wt1, bf16* __restrict__ Wt2) {
    int idx = blockIdx.x * blockDim.x + threadIdx.x;   // 0 .. 16383
    int o = idx >> 7, k = idx & 127;
    Wt1[o * 128 + k] = (bf16)W1[(size_t)k * 128 + o];
    Wt2[o * 128 + k] = (bf16)W2[(size_t)k * 128 + o];
}

// ---------------------------------------------------------------------------
// Kernel 1: KNN via WMMA f32 Gram tiles + per-row top-20 selection
//   block = 128 threads = 4 waves; each wave owns a 16-row tile.
// ---------------------------------------------------------------------------
__global__ void __launch_bounds__(128)
knn_kernel(const float* __restrict__ x, const float* __restrict__ sq,
           int* __restrict__ knn) {
    __shared__ float d2tile[4][16][16];
    __shared__ float listD[4][32][KNB];
    __shared__ int   listI[4][32][KNB];

    const int wave = threadIdx.x >> 5;
    const int lane = threadIdx.x & 31;
    const int half = lane >> 4;          // 0/1
    const int m    = lane & 15;          // 0..15

    const int rowTile  = blockIdx.x * 4 + wave;
    const int row_base = rowTile * 16;                 // global point id of row 0
    const int b        = row_base / NPTS;
    const int colStart = b * NPTS;                     // first global col of this cloud

    // Preload A fragments (16x64 f32, 16 k-steps) into registers.
    v2f a[16];
    {
        const float* arow = x + (size_t)(row_base + m) * CIN;
#pragma unroll
        for (int s = 0; s < 16; ++s) {
            a[s].x = arow[4 * s + 2 * half];
            a[s].y = arow[4 * s + 2 * half + 1];
        }
    }
    float si[8];
#pragma unroll
    for (int r = 0; r < 8; ++r) si[r] = sq[row_base + 8 * half + r];

#pragma unroll
    for (int k = 0; k < KNB; ++k) { listD[wave][lane][k] = 3.4e38f; listI[wave][lane][k] = colStart; }
    float curMax = 3.4e38f;
    int   curMaxPos = 0;

    for (int ct = 0; ct < NPTS / 16; ++ct) {
        const int col_base = colStart + ct * 16;       // global
        v8f acc = {};
        const float* brow = x + (size_t)(col_base + m) * CIN;
#pragma unroll
        for (int s = 0; s < 16; ++s) {
            v2f bf;
            bf.x = brow[4 * s + 2 * half];
            bf.y = brow[4 * s + 2 * half + 1];
            acc = __builtin_amdgcn_wmma_f32_16x16x4_f32(
                false, a[s], false, bf, (short)0, acc, false, false);
        }
        const float sj = sq[col_base + m];

        __syncthreads();
#pragma unroll
        for (int r = 0; r < 8; ++r)
            d2tile[wave][8 * half + r][m] = si[r] + sj - 2.0f * acc[r];
        __syncthreads();

#pragma unroll
        for (int c = 0; c < 8; ++c) {
            const int cc = 8 * half + c;
            const float d = d2tile[wave][m][cc];
            if (d < curMax) {
                listD[wave][lane][curMaxPos] = d;
                listI[wave][lane][curMaxPos] = col_base + cc;
                curMax = listD[wave][lane][0]; curMaxPos = 0;
#pragma unroll
                for (int k = 1; k < KNB; ++k) {
                    const float v = listD[wave][lane][k];
                    if (v > curMax) { curMax = v; curMaxPos = k; }
                }
            }
        }
    }
    __syncthreads();

    if (half == 0) {
        const int out_row = row_base + m;
        for (int t = 0; t < KNB; ++t) {
            float best = __builtin_inff();
            int bestPos = 0;
            for (int k = 0; k < 2 * KNB; ++k) {
                const int ln = m + (k >= KNB ? 16 : 0);
                const int kk = (k >= KNB) ? k - KNB : k;
                const float v = listD[wave][ln][kk];
                if (v < best) { best = v; bestPos = k; }
            }
            const int ln = m + (bestPos >= KNB ? 16 : 0);
            const int kk = (bestPos >= KNB) ? bestPos - KNB : bestPos;
            knn[(size_t)out_row * KNB + t] = listI[wave][ln][kk];
            listD[wave][ln][kk] = 3.4e38f;
        }
    }
}

// ---------------------------------------------------------------------------
// Kernel 2: edge MLP via bf16 WMMA 16x16x32 + LN + ReLU + max-K + residual
// ---------------------------------------------------------------------------
// A: bf16 LDS [rows][128]; B: bf16 global [out][in] (pre-transposed);
// acc f32; D layout: lane(half,m) holds rows 8*half+r, col m.
__device__ __forceinline__ void gemm_bf16(const bf16* Abuf, const bf16* __restrict__ Wt,
                                          const float* __restrict__ bias, float* Obuf,
                                          int wave, int half, int m) {
    for (int tt = wave; tt < 40; tt += 4) {     // 5 M-tiles x 8 N-tiles
        const int mt  = tt >> 3;
        const int nt  = tt & 7;
        const int col = nt * 16 + m;
        v8f acc = {};
#pragma unroll
        for (int s = 0; s < 4; ++s) {           // K = 128 in 4 steps of 32
            const int kb = 32 * s;
            // A 16x32 bf16: lanes 0-15 hold K kb..kb+15, lanes 16-31 hold K kb+16..kb+31
            v16bf af = *(const v16bf*)&Abuf[(mt * 16 + m) * COUT + kb + 16 * half];
            // B 32x16 bf16: lanes 0-15 hold K kb..kb+15, lanes 16-31 hold K kb+16..kb+31
            v16bf bfrag = *(const v16bf*)&Wt[(size_t)col * COUT + kb + 16 * half];
            acc = __builtin_amdgcn_wmma_f32_16x16x32_bf16(
                false, af, false, bfrag, (short)0, acc, false, false);
        }
        const float bcol = bias[col];
#pragma unroll
        for (int r = 0; r < 8; ++r)
            Obuf[(mt * 16 + 8 * half + r) * COUT + col] = acc[r] + bcol;
    }
}

__device__ __forceinline__ void ln_relu_to_bf16(const float* H, bf16* Obf,
                                                const float* __restrict__ g,
                                                const float* __restrict__ be, int tid) {
    if (tid < ROWS) {
        const float* row = H + tid * COUT;
        float s = 0.f, s2 = 0.f;
#pragma unroll 8
        for (int c = 0; c < COUT; ++c) { const float v = row[c]; s += v; s2 += v * v; }
        const float mean = s * (1.0f / COUT);
        const float var  = s2 * (1.0f / COUT) - mean * mean;
        const float rstd = rsqrtf(var + 1e-5f);
#pragma unroll 8
        for (int c = 0; c < COUT; ++c) {
            const float v = (row[c] - mean) * rstd * g[c] + be[c];
            Obf[tid * COUT + c] = (bf16)(v > 0.f ? v : 0.f);
        }
    }
}

__device__ __forceinline__ void ln_relu_inplace(float* H, const float* __restrict__ g,
                                                const float* __restrict__ be, int tid) {
    if (tid < ROWS) {
        float* row = H + tid * COUT;
        float s = 0.f, s2 = 0.f;
#pragma unroll 8
        for (int c = 0; c < COUT; ++c) { const float v = row[c]; s += v; s2 += v * v; }
        const float mean = s * (1.0f / COUT);
        const float var  = s2 * (1.0f / COUT) - mean * mean;
        const float rstd = rsqrtf(var + 1e-5f);
#pragma unroll 8
        for (int c = 0; c < COUT; ++c) {
            const float v = (row[c] - mean) * rstd * g[c] + be[c];
            row[c] = v > 0.f ? v : 0.f;
        }
    }
}

__global__ void __launch_bounds__(128)
edgeconv_mlp_kernel(const float* __restrict__ x, const int* __restrict__ knn,
                    const bf16* __restrict__ Wt1, const float* __restrict__ b1,
                    const float* __restrict__ g1, const float* __restrict__ be1,
                    const bf16* __restrict__ Wt2, const float* __restrict__ b2,
                    const float* __restrict__ g2, const float* __restrict__ be2,
                    const float* __restrict__ Wres, const float* __restrict__ bres,
                    float* __restrict__ out) {
    __shared__ bf16  Ebf[ROWS * COUT];    // 20 KB: bf16 A-matrix (E, then H1)
    __shared__ float Hf32[ROWS * COUT];   // 40 KB: f32 GEMM out / LN buffer
    float* stage = Hf32;                  // alias: raw xj staging (20 KB), dead before GEMM1

    const int tid  = threadIdx.x;
    const int wave = tid >> 5;
    const int lane = tid & 31;
    const int half = lane >> 4;
    const int m    = lane & 15;
    const int gbase = blockIdx.x * PG;

    __builtin_prefetch(Wt1 + tid * 128, 0, 0);   // global_prefetch_b8, L2 warm
    __builtin_prefetch(Wt2 + tid * 128, 0, 0);

    // Phase A1: async-stage neighbor rows xj into LDS (ASYNCcnt path).
    // 128 threads = 8 rows/pass x 16 lanes x 16B; 10 passes for 80 rows.
#pragma unroll 1
    for (int pass = 0; pass < ROWS / 8; ++pass) {
        const int r   = pass * 8 + (tid >> 4);        // 0..79
        const int l16 = tid & 15;
        const int p = r / KNB, k = r % KNB;
        const int gj = knn[(size_t)(gbase + p) * KNB + k];
        const float* gsrc = x + (size_t)gj * CIN + l16 * 4;
        unsigned ldsa = (unsigned)(uintptr_t)&stage[r * CIN + l16 * 4];
        asm volatile("global_load_async_to_lds_b128 %0, %1, off"
                     :: "v"(ldsa), "v"(gsrc) : "memory");
    }
    asm volatile("s_wait_asynccnt 0x0" ::: "memory");
    __syncthreads();

    // Phase A2: build bf16 E rows [xi | xj - xi]
    for (int r = wave; r < ROWS; r += 4) {
        const int p = r / KNB;
        const int gi = gbase + p;
        const float* xi = x + (size_t)gi * CIN;
        for (int c = lane; c < CIN; c += 32) {
            const float vi = xi[c];
            Ebf[r * COUT + c]       = (bf16)vi;
            Ebf[r * COUT + CIN + c] = (bf16)(stage[r * CIN + c] - vi);
        }
    }
    __syncthreads();

    // GEMM1 (bf16 WMMA) -> Hf32, LN+ReLU -> Ebf (bf16)
    gemm_bf16(Ebf, Wt1, b1, Hf32, wave, half, m);
    __syncthreads();
    ln_relu_to_bf16(Hf32, Ebf, g1, be1, tid);
    __syncthreads();

    // GEMM2 (bf16 WMMA) -> Hf32, LN+ReLU in place
    gemm_bf16(Ebf, Wt2, b2, Hf32, wave, half, m);
    __syncthreads();
    ln_relu_inplace(Hf32, g2, be2, tid);
    __syncthreads();

    // Phase D: max over K + residual projection
    for (int o = tid; o < PG * COUT; o += 128) {
        const int p = o >> 7, c = o & 127;
        float mx = -__builtin_inff();
#pragma unroll
        for (int k = 0; k < KNB; ++k)
            mx = fmaxf(mx, Hf32[(p * KNB + k) * COUT + c]);
        const int gi = gbase + p;
        const float* xi = x + (size_t)gi * CIN;
        float r = bres[c];
#pragma unroll 8
        for (int kk = 0; kk < CIN; ++kk) r += xi[kk] * Wres[(size_t)kk * COUT + c];
        out[(size_t)gi * COUT + c] = mx + r;
    }
}

// ---------------------------------------------------------------------------
// Launch
// ---------------------------------------------------------------------------
extern "C" void kernel_launch(void* const* d_in, const int* in_sizes, int n_in,
                              void* d_out, int out_size, void* d_ws, size_t ws_size,
                              hipStream_t stream) {
    const float* x    = (const float*)d_in[0];
    // d_in[1] = batch (int64) — implicit in contiguous layout, unused
    const float* W1   = (const float*)d_in[2];
    const float* b1   = (const float*)d_in[3];
    const float* g1   = (const float*)d_in[4];
    const float* be1  = (const float*)d_in[5];
    const float* W2   = (const float*)d_in[6];
    const float* b2   = (const float*)d_in[7];
    const float* g2   = (const float*)d_in[8];
    const float* be2  = (const float*)d_in[9];
    const float* Wres = (const float*)d_in[10];
    const float* bres = (const float*)d_in[11];
    float* out = (float*)d_out;

    char* ws = (char*)d_ws;
    float* sq  = (float*)ws;                                        // 128 KB
    int*   knn = (int*)(ws + (size_t)TOTAL * sizeof(float));        // 2.5 MB
    bf16*  Wt1 = (bf16*)(ws + (size_t)TOTAL * sizeof(float)
                            + (size_t)TOTAL * KNB * sizeof(int));   // 32 KB
    bf16*  Wt2 = Wt1 + 128 * 128;                                   // 32 KB

    sq_kernel<<<TOTAL / 256, 256, 0, stream>>>(x, sq);
    prep_weights<<<64, 256, 0, stream>>>(W1, W2, Wt1, Wt2);
    knn_kernel<<<TOTAL / 64, 128, 0, stream>>>(x, sq, knn);
    edgeconv_mlp_kernel<<<TOTAL / PG, 128, 0, stream>>>(
        x, knn, Wt1, b1, g1, be1, Wt2, b2, g2, be2, Wres, bres, out);
}
